// FreeTransAtt_24094766530794
// MI455X (gfx1250) — matrix operand order repaired and checked by reading, not compiled, not run on hardware
//
#include <hip/hip_runtime.h>
#include <hip/hip_bf16.h>
#include <math.h>

// ---------------------------------------------------------------------------
// AFT block for MI455X / gfx1250, wave32 + WMMA, register-blocked 4x along M.
//   q,k,v = input @ W{q,k,v}.T + b          (bf16 WMMA, f32 accum)
//   num   = exp(bias) @ (exp(k)*v)          (bf16 WMMA, f32 accum, per batch)
//   den   = exp(bias) @ exp(k)
//   out   = sigmoid(q) * num / den
// ---------------------------------------------------------------------------

typedef __attribute__((ext_vector_type(16))) __bf16 v16bf;
typedef __attribute__((ext_vector_type(8)))  __bf16 v8bf;
typedef __attribute__((ext_vector_type(8)))  float  v8f;
typedef __attribute__((ext_vector_type(4)))  float  v4f;

#define BS_  8
#define SEQ_ 1024
#define C_   1024

// Workspace layout (bytes)
#define OFF_INB   ((size_t)0)                         // bf16 input   [b][t][k]   16 MB
#define OFF_WQ    (OFF_INB  + (size_t)BS_*SEQ_*C_*2)  // bf16 Wq      [c][k]       2 MB
#define OFF_WK    (OFF_WQ   + (size_t)C_*C_*2)
#define OFF_WV    (OFF_WK   + (size_t)C_*C_*2)
#define OFF_EB    (OFF_WV   + (size_t)C_*C_*2)        // bf16 exp(bias) [t][s]     2 MB
#define OFF_QT    (OFF_EB   + (size_t)SEQ_*SEQ_*2)    // f32  qT      [b][c][t]   32 MB
#define OFF_EKT   (OFF_QT   + (size_t)BS_*SEQ_*C_*4)  // bf16 ekT     [b][c][s]   16 MB
#define OFF_EKVT  (OFF_EKT  + (size_t)BS_*SEQ_*C_*2)  // bf16 ekvT    [b][c][s]   16 MB

// ---------------------------------------------------------------------------
// Prep: f32 -> bf16 conversion (grid stride)
// ---------------------------------------------------------------------------
__global__ void cvt_bf16_kernel(const float* __restrict__ src,
                                __bf16* __restrict__ dst, int n) {
    int i = blockIdx.x * blockDim.x + threadIdx.x;
    int stride = gridDim.x * blockDim.x;
    for (; i < n; i += stride) dst[i] = (__bf16)src[i];
}

// Prep: bf16(exp(x))
__global__ void exp_bf16_kernel(const float* __restrict__ src,
                                __bf16* __restrict__ dst, int n) {
    int i = blockIdx.x * blockDim.x + threadIdx.x;
    int stride = gridDim.x * blockDim.x;
    for (; i < n; i += stride) dst[i] = (__bf16)__expf(src[i]);
}

// Fragment loaders (wave32): contiguous k = 16h + s, same order for A and B.
// A (16x32 bf16): lane half h: elems[0..15]=K k0+16h.. (contiguous, matches B).
__device__ inline v16bf load_a_frag(const __bf16* row, int k0, int half) {
    return *(const v16bf*)(row + k0 + 16 * half);
}
// B (32x16 bf16): lane half h, fixed column: 16 contiguous K at k0+16h.
// Single 32-byte vector load -> two b128 into consecutive VGPRs, no shuffles.
__device__ inline v16bf load_b_frag(const __bf16* col_row, int k0, int half) {
    return *(const v16bf*)(col_row + k0 + 16 * half);
}

// ---------------------------------------------------------------------------
// Fused QKV GEMM, 4 M-tiles (64 rows) x 16 cols per wave.
// Per K-step: 4 A frags + 3 B frags -> 12 WMMAs; weights amortized 4x.
// Writes qT (f32, [b][c][t]), ekT = bf16(exp(k)), ekvT = bf16(exp(k)*v)
// transposed so the AFT GEMM's B fragments are contiguous per lane.
// ---------------------------------------------------------------------------
__global__ __launch_bounds__(256)
void qkv_wmma_kernel(const __bf16* __restrict__ inb,
                     const __bf16* __restrict__ Wqb,
                     const __bf16* __restrict__ Wkb,
                     const __bf16* __restrict__ Wvb,
                     const float* __restrict__ bq,
                     const float* __restrict__ bk,
                     const float* __restrict__ bv,
                     float*  __restrict__ qT,
                     __bf16* __restrict__ ekT,
                     __bf16* __restrict__ ekvT) {
    const int wave = threadIdx.x >> 5;
    const int lane = threadIdx.x & 31;
    const int half = lane >> 4;
    const int lq   = lane & 15;

    // 8 batches x 16 m-blocks(64 rows) x 64 c-tiles = 8192 wave tiles
    const int wtile = blockIdx.x * 8 + wave;    // 0 .. 8191
    const int b     = wtile >> 10;
    const int rem   = wtile & 1023;
    const int t0    = (rem >> 6) << 6;          // m-block * 64
    const int c0    = (rem & 63) << 4;

    const __bf16* arow0 = inb + ((size_t)b * SEQ_ + t0 +  0 + lq) * C_;
    const __bf16* arow1 = inb + ((size_t)b * SEQ_ + t0 + 16 + lq) * C_;
    const __bf16* arow2 = inb + ((size_t)b * SEQ_ + t0 + 32 + lq) * C_;
    const __bf16* arow3 = inb + ((size_t)b * SEQ_ + t0 + 48 + lq) * C_;
    const __bf16* wqr   = Wqb + (size_t)(c0 + lq) * C_;
    const __bf16* wkr   = Wkb + (size_t)(c0 + lq) * C_;
    const __bf16* wvr   = Wvb + (size_t)(c0 + lq) * C_;

    v8f accq[4] = {{}, {}, {}, {}};
    v8f acck[4] = {{}, {}, {}, {}};
    v8f accv[4] = {{}, {}, {}, {}};

#pragma unroll 2
    for (int k0 = 0; k0 < C_; k0 += 32) {
        // prefetch next chunk of the streamed operands (global_prefetch_b8)
        if (k0 + 256 < C_) {
            __builtin_prefetch(arow0 + k0 + 256, 0, 1);
            __builtin_prefetch(arow2 + k0 + 256, 0, 1);
        }
        v16bf a0 = load_a_frag(arow0, k0, half);
        v16bf a1 = load_a_frag(arow1, k0, half);
        v16bf a2 = load_a_frag(arow2, k0, half);
        v16bf a3 = load_a_frag(arow3, k0, half);
        v16bf fq = load_b_frag(wqr, k0, half);
        v16bf fk = load_b_frag(wkr, k0, half);
        v16bf fv = load_b_frag(wvr, k0, half);

        accq[0] = __builtin_amdgcn_wmma_f32_16x16x32_bf16(false, a0, false, fq, (short)0, accq[0], false, false);
        accq[1] = __builtin_amdgcn_wmma_f32_16x16x32_bf16(false, a1, false, fq, (short)0, accq[1], false, false);
        accq[2] = __builtin_amdgcn_wmma_f32_16x16x32_bf16(false, a2, false, fq, (short)0, accq[2], false, false);
        accq[3] = __builtin_amdgcn_wmma_f32_16x16x32_bf16(false, a3, false, fq, (short)0, accq[3], false, false);
        acck[0] = __builtin_amdgcn_wmma_f32_16x16x32_bf16(false, a0, false, fk, (short)0, acck[0], false, false);
        acck[1] = __builtin_amdgcn_wmma_f32_16x16x32_bf16(false, a1, false, fk, (short)0, acck[1], false, false);
        acck[2] = __builtin_amdgcn_wmma_f32_16x16x32_bf16(false, a2, false, fk, (short)0, acck[2], false, false);
        acck[3] = __builtin_amdgcn_wmma_f32_16x16x32_bf16(false, a3, false, fk, (short)0, acck[3], false, false);
        accv[0] = __builtin_amdgcn_wmma_f32_16x16x32_bf16(false, a0, false, fv, (short)0, accv[0], false, false);
        accv[1] = __builtin_amdgcn_wmma_f32_16x16x32_bf16(false, a1, false, fv, (short)0, accv[1], false, false);
        accv[2] = __builtin_amdgcn_wmma_f32_16x16x32_bf16(false, a2, false, fv, (short)0, accv[2], false, false);
        accv[3] = __builtin_amdgcn_wmma_f32_16x16x32_bf16(false, a3, false, fv, (short)0, accv[3], false, false);
    }

    const float biasq = bq[c0 + lq];
    const float biask = bk[c0 + lq];
    const float biasv = bv[c0 + lq];

    // C-layout: lane column n = lq; VGPR r -> row m = r + 8*half (contiguous t)
    const size_t base = ((size_t)b * C_ + (c0 + lq)) * SEQ_ + t0 + 8 * half;

#pragma unroll
    for (int j = 0; j < 4; ++j) {
        const size_t off = base + j * 16;
        v4f qlo, qhi; v8bf e8, ev8;
#pragma unroll
        for (int r = 0; r < 8; ++r) {
            float qq = accq[j][r] + biasq;
            float ek = __expf(acck[j][r] + biask);
            float vv = accv[j][r] + biasv;
            if (r < 4) qlo[r] = qq; else qhi[r - 4] = qq;
            e8[r]  = (__bf16)ek;
            ev8[r] = (__bf16)(ek * vv);
        }
        *(v4f*)(qT + off)     = qlo;
        *(v4f*)(qT + off + 4) = qhi;
        *(v8bf*)(ekT + off)   = e8;
        *(v8bf*)(ekvT + off)  = ev8;
    }
}

// ---------------------------------------------------------------------------
// AFT GEMM, 4 t-tiles per wave; num/den share A frags, ekT/ekvT amortized 4x.
// Fused epilogue: out = sigmoid(q) * num / den (f32, reference layout).
// ---------------------------------------------------------------------------
__global__ __launch_bounds__(256)
void aft_wmma_kernel(const __bf16* __restrict__ ebb,
                     const __bf16* __restrict__ ekT,
                     const __bf16* __restrict__ ekvT,
                     const float*  __restrict__ qT,
                     float* __restrict__ out) {
    const int wave = threadIdx.x >> 5;
    const int lane = threadIdx.x & 31;
    const int half = lane >> 4;
    const int lq   = lane & 15;

    const int wtile = blockIdx.x * 8 + wave;    // 0 .. 8191
    const int b     = wtile >> 10;
    const int rem   = wtile & 1023;
    const int t0    = (rem >> 6) << 6;          // t-block * 64
    const int c0    = (rem & 63) << 4;

    const __bf16* arow0 = ebb + (size_t)(t0 +  0 + lq) * SEQ_;
    const __bf16* arow1 = ebb + (size_t)(t0 + 16 + lq) * SEQ_;
    const __bf16* arow2 = ebb + (size_t)(t0 + 32 + lq) * SEQ_;
    const __bf16* arow3 = ebb + (size_t)(t0 + 48 + lq) * SEQ_;
    const __bf16* nrow  = ekvT + ((size_t)b * C_ + c0 + lq) * SEQ_;
    const __bf16* drow  = ekT  + ((size_t)b * C_ + c0 + lq) * SEQ_;

    v8f accN[4] = {{}, {}, {}, {}};
    v8f accD[4] = {{}, {}, {}, {}};

#pragma unroll 2
    for (int k0 = 0; k0 < SEQ_; k0 += 32) {
        if (k0 + 256 < SEQ_) {
            __builtin_prefetch(nrow + k0 + 256, 0, 1);
            __builtin_prefetch(drow + k0 + 256, 0, 1);
        }
        v16bf a0 = load_a_frag(arow0, k0, half);
        v16bf a1 = load_a_frag(arow1, k0, half);
        v16bf a2 = load_a_frag(arow2, k0, half);
        v16bf a3 = load_a_frag(arow3, k0, half);
        v16bf fn = load_b_frag(nrow, k0, half);
        v16bf fd = load_b_frag(drow, k0, half);

        accN[0] = __builtin_amdgcn_wmma_f32_16x16x32_bf16(false, a0, false, fn, (short)0, accN[0], false, false);
        accN[1] = __builtin_amdgcn_wmma_f32_16x16x32_bf16(false, a1, false, fn, (short)0, accN[1], false, false);
        accN[2] = __builtin_amdgcn_wmma_f32_16x16x32_bf16(false, a2, false, fn, (short)0, accN[2], false, false);
        accN[3] = __builtin_amdgcn_wmma_f32_16x16x32_bf16(false, a3, false, fn, (short)0, accN[3], false, false);
        accD[0] = __builtin_amdgcn_wmma_f32_16x16x32_bf16(false, a0, false, fd, (short)0, accD[0], false, false);
        accD[1] = __builtin_amdgcn_wmma_f32_16x16x32_bf16(false, a1, false, fd, (short)0, accD[1], false, false);
        accD[2] = __builtin_amdgcn_wmma_f32_16x16x32_bf16(false, a2, false, fd, (short)0, accD[2], false, false);
        accD[3] = __builtin_amdgcn_wmma_f32_16x16x32_bf16(false, a3, false, fd, (short)0, accD[3], false, false);
    }

    const size_t qbase = ((size_t)b * C_ + (c0 + lq)) * SEQ_ + t0 + 8 * half;

#pragma unroll
    for (int j = 0; j < 4; ++j) {
        const size_t qoff = qbase + j * 16;
        v4f qlo = *(const v4f*)(qT + qoff);
        v4f qhi = *(const v4f*)(qT + qoff + 4);
#pragma unroll
        for (int r = 0; r < 8; ++r) {
            float qq  = (r < 4) ? qlo[r] : qhi[r - 4];
            float sig = 1.0f / (1.0f + __expf(-qq));
            float o   = sig * (accN[j][r] / accD[j][r]);
            out[((size_t)b * SEQ_ + t0 + j * 16 + 8 * half + r) * C_ + (c0 + lq)] = o;
        }
    }
}

// ---------------------------------------------------------------------------
extern "C" void kernel_launch(void* const* d_in, const int* in_sizes, int n_in,
                              void* d_out, int out_size, void* d_ws, size_t ws_size,
                              hipStream_t stream) {
    const float* input = (const float*)d_in[0];
    const float* Wq    = (const float*)d_in[1];
    const float* Wk    = (const float*)d_in[2];
    const float* Wv    = (const float*)d_in[3];
    const float* bq    = (const float*)d_in[4];
    const float* bk    = (const float*)d_in[5];
    const float* bv    = (const float*)d_in[6];
    const float* bias  = (const float*)d_in[7];
    float* out = (float*)d_out;

    char* ws = (char*)d_ws;
    __bf16* inb  = (__bf16*)(ws + OFF_INB);
    __bf16* Wqb  = (__bf16*)(ws + OFF_WQ);
    __bf16* Wkb  = (__bf16*)(ws + OFF_WK);
    __bf16* Wvb  = (__bf16*)(ws + OFF_WV);
    __bf16* ebb  = (__bf16*)(ws + OFF_EB);
    float*  qT   = (float*) (ws + OFF_QT);
    __bf16* ekT  = (__bf16*)(ws + OFF_EKT);
    __bf16* ekvT = (__bf16*)(ws + OFF_EKVT);

    // Prep conversions
    cvt_bf16_kernel<<<2048, 256, 0, stream>>>(input, inb, BS_ * SEQ_ * C_);
    cvt_bf16_kernel<<<1024, 256, 0, stream>>>(Wq, Wqb, C_ * C_);
    cvt_bf16_kernel<<<1024, 256, 0, stream>>>(Wk, Wkb, C_ * C_);
    cvt_bf16_kernel<<<1024, 256, 0, stream>>>(Wv, Wvb, C_ * C_);
    exp_bf16_kernel<<<1024, 256, 0, stream>>>(bias, ebb, SEQ_ * SEQ_);

    // Fused QKV GEMM: 8192 wave-tiles / 8 waves per block
    qkv_wmma_kernel<<<1024, 256, 0, stream>>>(inb, Wqb, Wkb, Wvb, bq, bk, bv,
                                              qT, ekT, ekvT);

    // AFT GEMM + fused epilogue: 8192 wave-tiles / 8 waves per block
    aft_wmma_kernel<<<1024, 256, 0, stream>>>(ebb, ekT, ekvT, qT, out);
}
